// MechanismAttention_56899726737906
// MI455X (gfx1250) — matrix operand order repaired and checked by reading, not compiled, not run on hardware
//
#include <hip/hip_runtime.h>
#include <hip/hip_bf16.h>
#include <math.h>

#define B_SZ     2
#define S_SZ     2048
#define DMODEL   1024
#define NHEADS   16
#define DHEAD    64
#define HMID     512
#define SHARP_F  2.0f

typedef __bf16 bf16_t;
typedef __attribute__((ext_vector_type(16))) __bf16 v16bf;
typedef __attribute__((ext_vector_type(8)))  __bf16 v8bf;
typedef __attribute__((ext_vector_type(8)))  float  v8f;

static __device__ __forceinline__ bf16_t f2bf(float f) {
    union { float f; unsigned u; } a; a.f = f;
    unsigned r = a.u + 0x7FFFu + ((a.u >> 16) & 1u);   // round-to-nearest-even
    unsigned short h = (unsigned short)(r >> 16);
    return __builtin_bit_cast(bf16_t, h);
}

static __device__ __forceinline__ v16bf load_afrag(const bf16_t* p) {
    // 16-bit A-frag: lane<16 holds K {0..7,16..23}, lane>=16 holds K {8..15,24..31}
    v8bf lo = *(const v8bf*)p;
    v8bf hi = *(const v8bf*)(p + 16);
    v16bf r;
#pragma unroll
    for (int e = 0; e < 8; ++e) { r[e] = lo[e]; r[e + 8] = hi[e]; }
    return r;
}

static __device__ __forceinline__ v16bf load_bfrag(const bf16_t* p) {
    // 16-bit B-frag: lane<16 holds K {0..7,16..23}, lane>=16 holds K {8..15,24..31} of its column's [N,K] row
    v8bf lo = *(const v8bf*)p;
    v8bf hi = *(const v8bf*)(p + 16);
    v16bf r;
#pragma unroll
    for (int e = 0; e < 8; ++e) { r[e] = lo[e]; r[e + 8] = hi[e]; }
    return r;
}

// ---------------------------------------------------------------- converts
__global__ __launch_bounds__(256)
void cvt_f32_bf16(const float* __restrict__ in, bf16_t* __restrict__ out, int n) {
    int i = blockIdx.x * 256 + threadIdx.x;
    if (i < n) out[i] = f2bf(in[i]);
}

// ---------------------------------------------------------------- generic WMMA GEMM
// C[M,N] = (A[M,K]_bf16 · W[N,K]_bf16^T + bias) * alpha  (+ optional exact GELU)
// mode: 0 = f32 row-major, 1 = bf16 row-major, 2 = bf16 V-transposed [B][H][64][S]
// block = 8 waves; wave tile 32x64 (2x4 WMMA tiles); block tile 128x128
__global__ __launch_bounds__(256)
void gemm_bf16_wmma(const bf16_t* __restrict__ A, const bf16_t* __restrict__ W,
                    const float* __restrict__ bias, void* __restrict__ out,
                    int M, int N, int Kd, int mode, float alpha, int act_gelu)
{
    const int lane = threadIdx.x & 31;
    const int wave = threadIdx.x >> 5;
    const int lrow = lane & 15;
    const int lhi  = lane >> 4;
    const int mw = blockIdx.y * 128 + (wave >> 1) * 32;
    const int nw = blockIdx.x * 128 + (wave & 1)  * 64;

    v8f acc[2][4] = {};
    for (int k = 0; k < Kd; k += 32) {
        v16bf a[2], b[4];
#pragma unroll
        for (int i = 0; i < 2; ++i)
            a[i] = load_afrag(A + (size_t)(mw + i * 16 + lrow) * Kd + k + lhi * 8);
#pragma unroll
        for (int j = 0; j < 4; ++j)
            b[j] = load_bfrag(W + (size_t)(nw + j * 16 + lrow) * Kd + k + lhi * 8);
#pragma unroll
        for (int i = 0; i < 2; ++i)
#pragma unroll
            for (int j = 0; j < 4; ++j)
                acc[i][j] = __builtin_amdgcn_wmma_f32_16x16x32_bf16(
                    false, a[i], false, b[j], (short)0, acc[i][j], false, false);
    }

#pragma unroll
    for (int i = 0; i < 2; ++i)
#pragma unroll
        for (int j = 0; j < 4; ++j)
#pragma unroll
            for (int e = 0; e < 8; ++e) {
                int row = mw + i * 16 + e + 8 * lhi;
                int col = nw + j * 16 + lrow;
                float v = (acc[i][j][e] + bias[col]) * alpha;
                if (act_gelu) v = 0.5f * v * (1.0f + erff(v * 0.70710678118f));
                if (mode == 0) {
                    ((float*)out)[(size_t)row * N + col] = v;
                } else if (mode == 1) {
                    ((bf16_t*)out)[(size_t)row * N + col] = f2bf(v);
                } else {
                    int bb = row >> 11, s = row & (S_SZ - 1);
                    int hh = col >> 6,  dd = col & (DHEAD - 1);
                    ((bf16_t*)out)[(((size_t)(bb * NHEADS + hh)) * DHEAD + dd) * S_SZ + s] = f2bf(v);
                }
            }
}

// ---------------------------------------------------------------- mech logits (hmid[4096,512]·Wm2 + bm2)
__global__ __launch_bounds__(256)
void mech_logits_kernel(const float* __restrict__ hmid, const float* __restrict__ Wm2,
                        const float* __restrict__ bm2, float* __restrict__ logits, int rows)
{
    int lane = threadIdx.x & 31;
    int row  = blockIdx.x * 8 + (threadIdx.x >> 5);
    if (row >= rows) return;
    float acc = 0.f;
    for (int i = lane; i < HMID; i += 32)
        acc += hmid[(size_t)row * HMID + i] * Wm2[i];
#pragma unroll
    for (int xm = 16; xm >= 1; xm >>= 1) acc += __shfl_xor(acc, xm, 32);
    if (lane == 0) logits[row] = acc + bm2[0];
}

// ---------------------------------------------------------------- mech softmax over S per batch
__global__ __launch_bounds__(256)
void mech_softmax_kernel(const float* __restrict__ logits,
                         float* __restrict__ mech, float* __restrict__ out_mech)
{
    int b = blockIdx.x, tid = threadIdx.x;
    __shared__ float red[256];
    float mx = -1e30f;
    for (int s = tid; s < S_SZ; s += 256) mx = fmaxf(mx, logits[b * S_SZ + s]);
    red[tid] = mx; __syncthreads();
    for (int w = 128; w > 0; w >>= 1) {
        if (tid < w) red[tid] = fmaxf(red[tid], red[tid + w]);
        __syncthreads();
    }
    mx = red[0]; __syncthreads();
    float sm = 0.f;
    for (int s = tid; s < S_SZ; s += 256) sm += __expf(logits[b * S_SZ + s] - mx);
    red[tid] = sm; __syncthreads();
    for (int w = 128; w > 0; w >>= 1) {
        if (tid < w) red[tid] += red[tid + w];
        __syncthreads();
    }
    float inv = 1.f / red[0];
    for (int s = tid; s < S_SZ; s += 256) {
        float v = __expf(logits[b * S_SZ + s] - mx) * inv;
        mech[b * S_SZ + s] = v;
        out_mech[b * S_SZ + s] = v;
    }
}

// ---------------------------------------------------------------- flash attention
// block = 4 waves = 128 queries of one (b,h); each wave owns 32 queries.
// K/V 32-key chunks double-buffered in LDS via async-to-LDS copies (ASYNCcnt).
__global__ __launch_bounds__(128)
void flash_attn_kernel(const bf16_t* __restrict__ Q, const bf16_t* __restrict__ K,
                       const bf16_t* __restrict__ Vt, const float* __restrict__ mech,
                       const int* __restrict__ mask, bf16_t* __restrict__ ctx)
{
    const int tid  = threadIdx.x;
    const int lane = tid & 31;
    const int wave = tid >> 5;
    const int lrow = lane & 15;
    const int lhi  = lane >> 4;

    int t = blockIdx.x;
    const int nqb = S_SZ / 128;
    const int qb  = (t % nqb) * 128; t /= nqb;
    const int h   = t % NHEADS;
    const int b   = t / NHEADS;
    const int qw  = qb + wave * 32;
    const int bh  = b * NHEADS + h;

    // LDS: K[2][32][64] | V[2][64][32] | P[4 waves][32][32]   (elems, bf16)
    __shared__ __align__(16) bf16_t smem[2 * 2048 + 2 * 2048 + 4 * 1024];
    const unsigned lds0 = (unsigned)(size_t)(&smem[0]);
    bf16_t* pl = &smem[8192 + wave * 1024];

    // Q fragments: 2 query tiles x 2 K-dim steps (Q pre-scaled by 1/sqrt(Dh))
    v16bf qa[2][2];
#pragma unroll
    for (int qi = 0; qi < 2; ++qi)
#pragma unroll
        for (int d = 0; d < 2; ++d)
            qa[qi][d] = load_afrag(Q + (size_t)(b * S_SZ + qw + qi * 16 + lrow) * DMODEL
                                     + h * DHEAD + d * 32 + lhi * 8);

    v8f o[2][4] = {};
    v8f mrun[2], lrun[2];
#pragma unroll
    for (int qi = 0; qi < 2; ++qi)
#pragma unroll
        for (int e = 0; e < 8; ++e) { mrun[qi][e] = -1e30f; lrun[qi][e] = 0.f; }

    // async stage one 32-key chunk (K: 32x64, V: 64x32) into LDS buffer `buf`
    auto issue_chunk = [&](int kc, int buf) {
        const unsigned kof = buf ? 2048u : 0u;
        const unsigned vof = buf ? 6144u : 4096u;
#pragma unroll
        for (int r = 0; r < 2; ++r) {
            int c = tid + r * 128;
            {   // K rows: 64 elems = 8 x 16B parts
                int key = c >> 3, part = c & 7;
                const bf16_t* g = K + (size_t)(b * S_SZ + kc + key) * DMODEL + h * DHEAD + part * 8;
                unsigned l = lds0 + (kof + (unsigned)(key * 64 + part * 8)) * 2u;
                asm volatile("global_load_async_to_lds_b128 %0, %1, off"
                             :: "v"(l), "v"((unsigned long long)(size_t)g) : "memory");
            }
            {   // V rows: 32 elems = 4 x 16B parts
                int d = c >> 2, part = c & 3;
                const bf16_t* g = Vt + ((size_t)bh * DHEAD + d) * S_SZ + kc + part * 8;
                unsigned l = lds0 + (vof + (unsigned)(d * 32 + part * 8)) * 2u;
                asm volatile("global_load_async_to_lds_b128 %0, %1, off"
                             :: "v"(l), "v"((unsigned long long)(size_t)g) : "memory");
            }
        }
    };

    issue_chunk(0, 0);
    for (int kc = 0; kc < S_SZ; kc += 32) {
        const int cur = (kc >> 5) & 1;
        if (kc + 32 < S_SZ) {
            issue_chunk(kc + 32, cur ^ 1);                    // prefetch next chunk
            asm volatile("s_wait_asynccnt 0x4" ::: "memory"); // retire current chunk
        } else {
            asm volatile("s_wait_asynccnt 0x0" ::: "memory");
        }
        __syncthreads();

        const bf16_t* kl = &smem[cur ? 2048 : 0];
        const bf16_t* vl = &smem[cur ? 6144 : 4096];

        // scores: 2 query tiles x 2 key tiles
        v8f s[2][2] = {};
#pragma unroll
        for (int d = 0; d < 2; ++d) {
            v16bf kb[2];
#pragma unroll
            for (int kj = 0; kj < 2; ++kj)
                kb[kj] = load_bfrag(kl + (kj * 16 + lrow) * 64 + d * 32 + lhi * 8);
#pragma unroll
            for (int qi = 0; qi < 2; ++qi)
#pragma unroll
                for (int kj = 0; kj < 2; ++kj)
                    s[qi][kj] = __builtin_amdgcn_wmma_f32_16x16x32_bf16(
                        false, qa[qi][d], false, kb[kj], (short)0, s[qi][kj], false, false);
        }

        const int key0 = kc + lrow, key1 = kc + 16 + lrow;
        const int msk0 = mask[b * S_SZ + key0], msk1 = mask[b * S_SZ + key1];
        const float add0 = SHARP_F * mech[b * S_SZ + key0];
        const float add1 = SHARP_F * mech[b * S_SZ + key1];

#pragma unroll
        for (int qi = 0; qi < 2; ++qi) {
            v8f p0, p1;
#pragma unroll
            for (int e = 0; e < 8; ++e) {
                p0[e] = msk0 ? s[qi][0][e] + add0 : -1e30f;
                p1[e] = msk1 ? s[qi][1][e] + add1 : -1e30f;
            }
            v8f tm;
#pragma unroll
            for (int e = 0; e < 8; ++e) tm[e] = fmaxf(p0[e], p1[e]);
#pragma unroll
            for (int xm = 1; xm < 16; xm <<= 1)
#pragma unroll
                for (int e = 0; e < 8; ++e) tm[e] = fmaxf(tm[e], __shfl_xor(tm[e], xm, 32));
            v8f scl, rs;
#pragma unroll
            for (int e = 0; e < 8; ++e) {
                float mn = fmaxf(mrun[qi][e], tm[e]);
                scl[e]   = __expf(mrun[qi][e] - mn);
                p0[e]    = __expf(p0[e] - mn);
                p1[e]    = __expf(p1[e] - mn);
                rs[e]    = p0[e] + p1[e];
                mrun[qi][e] = mn;
            }
#pragma unroll
            for (int xm = 1; xm < 16; xm <<= 1)
#pragma unroll
                for (int e = 0; e < 8; ++e) rs[e] += __shfl_xor(rs[e], xm, 32);
#pragma unroll
            for (int e = 0; e < 8; ++e) lrun[qi][e] = lrun[qi][e] * scl[e] + rs[e];
#pragma unroll
            for (int t4 = 0; t4 < 4; ++t4)
#pragma unroll
                for (int e = 0; e < 8; ++e) o[qi][t4][e] *= scl[e];
            // stage P (C-layout -> A-layout), per-wave private LDS slab
#pragma unroll
            for (int e = 0; e < 8; ++e) {
                pl[(qi * 16 + e + 8 * lhi) * 32 + lrow]      = f2bf(p0[e]);
                pl[(qi * 16 + e + 8 * lhi) * 32 + 16 + lrow] = f2bf(p1[e]);
            }
        }
        asm volatile("s_wait_dscnt 0x0" ::: "memory");  // wave-internal DS ordering

        v16bf vb[4];
#pragma unroll
        for (int t4 = 0; t4 < 4; ++t4)
            vb[t4] = load_bfrag(vl + (t4 * 16 + lrow) * 32 + lhi * 8);
#pragma unroll
        for (int qi = 0; qi < 2; ++qi) {
            v16bf pa = load_afrag(pl + (qi * 16 + lrow) * 32 + lhi * 8);
#pragma unroll
            for (int t4 = 0; t4 < 4; ++t4)
                o[qi][t4] = __builtin_amdgcn_wmma_f32_16x16x32_bf16(
                    false, pa, false, vb[t4], (short)0, o[qi][t4], false, false);
        }
        __syncthreads();   // all waves done reading before next async overwrite
    }

#pragma unroll
    for (int qi = 0; qi < 2; ++qi)
#pragma unroll
        for (int t4 = 0; t4 < 4; ++t4)
#pragma unroll
            for (int e = 0; e < 8; ++e) {
                float v = o[qi][t4][e] / lrun[qi][e];
                ctx[(size_t)(b * S_SZ + qw + qi * 16 + e + 8 * lhi) * DMODEL
                    + h * DHEAD + t4 * 16 + lrow] = f2bf(v);
            }
}

// ---------------------------------------------------------------- host glue
extern "C" void kernel_launch(void* const* d_in, const int* in_sizes, int n_in,
                              void* d_out, int out_size, void* d_ws, size_t ws_size,
                              hipStream_t stream) {
    const float* x   = (const float*)d_in[0];
    const int*   msk = (const int*)d_in[1];
    const float* Wq  = (const float*)d_in[2];
    const float* bq  = (const float*)d_in[3];
    const float* Wk  = (const float*)d_in[4];
    const float* bk  = (const float*)d_in[5];
    const float* Wv  = (const float*)d_in[6];
    const float* bv  = (const float*)d_in[7];
    const float* Wo  = (const float*)d_in[8];
    const float* bo  = (const float*)d_in[9];
    const float* Wm1 = (const float*)d_in[10];
    const float* bm1 = (const float*)d_in[11];
    const float* Wm2 = (const float*)d_in[12];
    const float* bm2 = (const float*)d_in[13];
    float* out = (float*)d_out;

    const int M  = B_SZ * S_SZ;           // 4096
    char* ws = (char*)d_ws;
    size_t off = 0;
    auto alloc = [&](size_t bytes) -> void* {
        void* p = ws + off;
        off = (off + bytes + 255) & ~(size_t)255;
        return p;
    };
    bf16_t* xb    = (bf16_t*)alloc((size_t)M * DMODEL * 2);
    bf16_t* qb    = (bf16_t*)alloc((size_t)M * DMODEL * 2);
    bf16_t* kb    = (bf16_t*)alloc((size_t)M * DMODEL * 2);
    bf16_t* vtb   = (bf16_t*)alloc((size_t)B_SZ * NHEADS * DHEAD * S_SZ * 2);
    bf16_t* ctxb  = (bf16_t*)alloc((size_t)M * DMODEL * 2);
    float*  hmid  = (float*) alloc((size_t)M * HMID * 4);
    bf16_t* wqb   = (bf16_t*)alloc((size_t)DMODEL * DMODEL * 2);
    bf16_t* wkb   = (bf16_t*)alloc((size_t)DMODEL * DMODEL * 2);
    bf16_t* wvb   = (bf16_t*)alloc((size_t)DMODEL * DMODEL * 2);
    bf16_t* wob   = (bf16_t*)alloc((size_t)DMODEL * DMODEL * 2);
    bf16_t* wm1b  = (bf16_t*)alloc((size_t)HMID * DMODEL * 2);
    float*  logit = (float*) alloc((size_t)M * 4);
    float*  mech  = (float*) alloc((size_t)M * 4);

    // bf16 conversions (Linear weights stay [N,K]: already WMMA B-frag layout)
    int nx = M * DMODEL;
    cvt_f32_bf16<<<(nx + 255) / 256, 256, 0, stream>>>(x, xb, nx);
    int nw = DMODEL * DMODEL;
    cvt_f32_bf16<<<(nw + 255) / 256, 256, 0, stream>>>(Wq, wqb, nw);
    cvt_f32_bf16<<<(nw + 255) / 256, 256, 0, stream>>>(Wk, wkb, nw);
    cvt_f32_bf16<<<(nw + 255) / 256, 256, 0, stream>>>(Wv, wvb, nw);
    cvt_f32_bf16<<<(nw + 255) / 256, 256, 0, stream>>>(Wo, wob, nw);
    int nm1 = HMID * DMODEL;
    cvt_f32_bf16<<<(nm1 + 255) / 256, 256, 0, stream>>>(Wm1, wm1b, nm1);

    // projections: Q scaled by 1/sqrt(Dh); V stored transposed [B][H][64][S]
    dim3 blk(256);
    dim3 gP(DMODEL / 128, M / 128);
    gemm_bf16_wmma<<<gP, blk, 0, stream>>>(xb, wqb, bq, qb,  M, DMODEL, DMODEL, 1, 0.125f, 0);
    gemm_bf16_wmma<<<gP, blk, 0, stream>>>(xb, wkb, bk, kb,  M, DMODEL, DMODEL, 1, 1.0f,   0);
    gemm_bf16_wmma<<<gP, blk, 0, stream>>>(xb, wvb, bv, vtb, M, DMODEL, DMODEL, 2, 1.0f,   0);

    // mechanism MLP
    dim3 gM(HMID / 128, M / 128);
    gemm_bf16_wmma<<<gM, blk, 0, stream>>>(xb, wm1b, bm1, hmid, M, HMID, DMODEL, 0, 1.0f, 1);
    mech_logits_kernel<<<M / 8, 256, 0, stream>>>(hmid, Wm2, bm2, logit, M);
    mech_softmax_kernel<<<B_SZ, 256, 0, stream>>>(logit, mech, out + (size_t)M * DMODEL);

    // flash attention: 512 blocks x 4 waves, async double-buffered K/V
    flash_attn_kernel<<<B_SZ * NHEADS * (S_SZ / 128), 128, 0, stream>>>(qb, kb, vtb, mech, msk, ctxb);

    // output projection -> d_out (f32)
    gemm_bf16_wmma<<<gP, blk, 0, stream>>>(ctxb, wob, bo, out, M, DMODEL, DMODEL, 0, 1.0f, 0);

    (void)in_sizes; (void)n_in; (void)out_size; (void)ws_size;
}
